// ProductAttentionLayer_45724221833754
// MI455X (gfx1250) — matrix operand order repaired and checked by reading, not compiled, not run on hardware
//
#include <hip/hip_runtime.h>
#include <hip/hip_bf16.h>
#include <stdint.h>

typedef __attribute__((ext_vector_type(16))) __bf16 v16bf;
typedef __attribute__((ext_vector_type(8)))  __bf16 v8bf;
typedef __attribute__((ext_vector_type(8)))  float  v8f;
typedef __attribute__((ext_vector_type(8)))  unsigned int v8u;

#define B_    8
#define LQ_   2048
#define LK_   2048
#define DH_   128
#define QTILE 128   // q rows per workgroup (16 per wave x 8 waves)
#define KTILE 32    // keys per iteration

// ---- bf16 pack: probe the hardware packed convert directly ----
static __device__ __forceinline__ unsigned int packbf(float lo, float hi) {
  unsigned int r;
  asm("v_cvt_pk_bf16_f32 %0, %1, %2" : "=v"(r) : "v"(lo), "v"(hi));
  return r;
}

#if __has_builtin(__builtin_amdgcn_rcpf)
static __device__ __forceinline__ float fast_rcp(float x) {
  return __builtin_amdgcn_rcpf(x);
}
#else
static __device__ __forceinline__ float fast_rcp(float x) { return 1.0f / x; }
#endif

static __device__ __forceinline__ v16bf cat8(v8bf a, v8bf b) {
  return __builtin_shufflevector(a, b, 0,1,2,3,4,5,6,7,8,9,10,11,12,13,14,15);
}

__global__ __launch_bounds__(256)
void sigmoid_attn_kernel(const float* __restrict__ Q,
                         const float* __restrict__ K,
                         const float* __restrict__ V,
                         const int*   __restrict__ M,
                         float* __restrict__ O) {
  // K tile row-major [key][d], V tile transposed [d][key], both bf16.
  __shared__ __align__(16) unsigned short ldsK [KTILE * DH_];
  __shared__ __align__(16) unsigned short ldsVT[DH_ * KTILE];

  const int tid  = threadIdx.x;
  const int wave = tid >> 5;
  const int lane = tid & 31;
  const int bi   = blockIdx.y;
  const int qb   = blockIdx.x * QTILE + wave * 16;
  const int hi   = (lane >> 4) & 1;     // lane half: 0 -> lanes 0-15, 1 -> 16-31
  const int col  = lane & 15;

  const float scale = 0.08838834764831845f;  // 1/sqrt(128), folded into Q
  const float negv  = -100000.0f;

  // ---- Hoisted: (Q * scale)^T as WMMA B operand (32x16), per 32-d chunk.
  // B layout: lane L holds column q = L%16; lanes 0-15 hold K(d)={0..7,16..23}
  // packed 2/VGPR, lanes 16-31 hold K(d)={8..15,24..31}.
  const int qrow = qb + col;
  const float* qptr = Q + ((size_t)bi * LQ_ + qrow) * DH_;
  v16bf qf[4];
#pragma unroll
  for (int c = 0; c < 4; ++c) {
    const float4* p = (const float4*)(qptr + c * 32 + hi * 8);
    float4 x0 = p[0], x1 = p[1], x2 = p[4], x3 = p[5];
    v8u t;
    t[0] = packbf(x0.x * scale, x0.y * scale);
    t[1] = packbf(x0.z * scale, x0.w * scale);
    t[2] = packbf(x1.x * scale, x1.y * scale);
    t[3] = packbf(x1.z * scale, x1.w * scale);
    t[4] = packbf(x2.x * scale, x2.y * scale);
    t[5] = packbf(x2.z * scale, x2.w * scale);
    t[6] = packbf(x3.x * scale, x3.y * scale);
    t[7] = packbf(x3.z * scale, x3.w * scale);
    qf[c] = __builtin_bit_cast(v16bf, t);
  }

  // ---- O^T accumulators: 8 tiles of 16(d) x 16(q), f32
  v8f ot[8];
#pragma unroll
  for (int t = 0; t < 8; ++t) ot[t] = (v8f){0.f,0.f,0.f,0.f,0.f,0.f,0.f,0.f};

  // staging index maps
  const int srow = tid >> 3;            // K staging: key row 0..31
  const int scol = (tid & 7) << 4;      // K staging: 16-wide d chunk
  const int vkp  = (tid >> 4) << 1;     // V staging: even key 0..30 (pair)
  const int vdg  = (tid & 15) << 3;     // V staging: 8-wide d group
  const float* kbase = K + (size_t)bi * LK_ * DH_;
  const float* vbase = V + (size_t)bi * LK_ * DH_;
  const int*   mrow  = M + ((size_t)bi * LQ_ + qrow) * (size_t)LK_;

  for (int kb = 0; kb < LK_; kb += KTILE) {
    __syncthreads();
    { // stage K tile: fp32 -> bf16, row-major
      const float4* p = (const float4*)(kbase + (size_t)(kb + srow) * DH_ + scol);
      float4 x0 = p[0], x1 = p[1], x2 = p[2], x3 = p[3];
      unsigned int* d = (unsigned int*)&ldsK[srow * DH_ + scol];
      d[0]=packbf(x0.x,x0.y); d[1]=packbf(x0.z,x0.w);
      d[2]=packbf(x1.x,x1.y); d[3]=packbf(x1.z,x1.w);
      d[4]=packbf(x2.x,x2.y); d[5]=packbf(x2.z,x2.w);
      d[6]=packbf(x3.x,x3.y); d[7]=packbf(x3.z,x3.w);
    }
    { // stage V tile transposed: ldsVT[d][key], two keys packed per b32 store
      const float4* pa = (const float4*)(vbase + (size_t)(kb + vkp    ) * DH_ + vdg);
      const float4* pb = (const float4*)(vbase + (size_t)(kb + vkp + 1) * DH_ + vdg);
      float4 a0 = pa[0], a1 = pa[1];
      float4 b0 = pb[0], b1 = pb[1];
      float av[8] = {a0.x,a0.y,a0.z,a0.w, a1.x,a1.y,a1.z,a1.w};
      float bv[8] = {b0.x,b0.y,b0.z,b0.w, b1.x,b1.y,b1.z,b1.w};
#pragma unroll
      for (int j = 0; j < 8; ++j)
        *(unsigned int*)&ldsVT[(vdg + j) * KTILE + vkp] = packbf(av[j], bv[j]);
    }
    __syncthreads();

    // ---- S^T = K_tile x (Q*scale)^T : two 16(key)x16(q) accum tiles
    v8f st0 = (v8f){0.f,0.f,0.f,0.f,0.f,0.f,0.f,0.f};
    v8f st1 = st0;
#pragma unroll
    for (int c = 0; c < 4; ++c) {
      // A layout (16x32 bf16): lane row M=key=col; d = c*32 + {0..7,16..23}
      // for lanes 0-15, {8..15,24..31} for lanes 16-31.
      const unsigned short* k0 = &ldsK[col * DH_        + c * 32 + hi * 8];
      const unsigned short* k1 = &ldsK[(16 + col) * DH_ + c * 32 + hi * 8];
      v16bf a0 = cat8(*(const v8bf*)(k0), *(const v8bf*)(k0 + 16));
      v16bf a1 = cat8(*(const v8bf*)(k1), *(const v8bf*)(k1 + 16));
      st0 = __builtin_amdgcn_wmma_f32_16x16x32_bf16(false, a0, false, qf[c],
                                                    (short)0, st0, false, false);
      st1 = __builtin_amdgcn_wmma_f32_16x16x32_bf16(false, a1, false, qf[c],
                                                    (short)0, st1, false, false);
    }

    // ---- additive mask (contiguous per lane thanks to S^T layout) + sigmoid
    const int4* mp0 = (const int4*)(mrow + kb + hi * 8);
    const int4* mp1 = (const int4*)(mrow + kb + 16 + hi * 8);
    int4 m00 = mp0[0], m01 = mp0[1];
    int4 m10 = mp1[0], m11 = mp1[1];
    if (kb + KTILE < LK_) __builtin_prefetch(mrow + kb + KTILE, 0, 0);

    int mm0[8] = {m00.x,m00.y,m00.z,m00.w, m01.x,m01.y,m01.z,m01.w};
    int mm1[8] = {m10.x,m10.y,m10.z,m10.w, m11.x,m11.y,m11.z,m11.w};
    float p0[8], p1[8];
#pragma unroll
    for (int v = 0; v < 8; ++v) {
      float s0 = st0[v] + (float)mm0[v] * negv;   // scale already folded into Q
      float s1 = st1[v] + (float)mm1[v] * negv;
      p0[v] = fast_rcp(1.0f + __expf(-s0));
      p1[v] = fast_rcp(1.0f + __expf(-s1));
    }
    unsigned int pk0[4], pk1[4];
#pragma unroll
    for (int j = 0; j < 4; ++j) {
      pk0[j] = packbf(p0[2*j], p0[2*j+1]);
      pk1[j] = packbf(p1[2*j], p1[2*j+1]);
    }

    // ---- assemble P^T as B operand (32 keys x 16 q): lane<16 needs keys
    // 0..15 (own st0 keys 0..7 + partner(L+16) st0 keys 8..15); lane>=16
    // needs keys 16..31. Lane xor-16 swap via ds_swizzle (SWAPX16 = 0x401F).
    v8u bpw;
#pragma unroll
    for (int j = 0; j < 4; ++j) {
      unsigned int sw0 = (unsigned int)__builtin_amdgcn_ds_swizzle((int)pk0[j], 0x401F);
      unsigned int sw1 = (unsigned int)__builtin_amdgcn_ds_swizzle((int)pk1[j], 0x401F);
      bpw[j]     = hi ? sw1    : pk0[j];
      bpw[4 + j] = hi ? pk1[j] : sw0;
    }
    v16bf bp = __builtin_bit_cast(v16bf, bpw);

    // ---- O^T += V^T x P^T : 8 d-tiles of 16x16, in two groups of 4 so the
    // DS fragment loads clause together ahead of the WMMA chain.
#pragma unroll
    for (int g = 0; g < 2; ++g) {
      v16bf av[4];
#pragma unroll
      for (int t = 0; t < 4; ++t) {
        const unsigned short* vp = &ldsVT[((g * 4 + t) * 16 + col) * KTILE + hi * 16];
        av[t] = cat8(*(const v8bf*)(vp), *(const v8bf*)(vp + 8));
      }
#pragma unroll
      for (int t = 0; t < 4; ++t) {
        ot[g * 4 + t] = __builtin_amdgcn_wmma_f32_16x16x32_bf16(
            false, av[t], false, bp, (short)0, ot[g * 4 + t], false, false);
      }
    }
  }

  // ---- store O: lane col -> q row; tile t rows are d = t*16 + hi*8 + v
  float* orow = O + ((size_t)bi * LQ_ + qrow) * DH_;
#pragma unroll
  for (int t = 0; t < 8; ++t) {
    float4 lo4 = make_float4(ot[t][0], ot[t][1], ot[t][2], ot[t][3]);
    float4 hi4 = make_float4(ot[t][4], ot[t][5], ot[t][6], ot[t][7]);
    *(float4*)(orow + t * 16 + hi * 8 + 0) = lo4;
    *(float4*)(orow + t * 16 + hi * 8 + 4) = hi4;
  }
}

extern "C" void kernel_launch(void* const* d_in, const int* in_sizes, int n_in,
                              void* d_out, int out_size, void* d_ws, size_t ws_size,
                              hipStream_t stream) {
  (void)in_sizes; (void)n_in; (void)out_size; (void)d_ws; (void)ws_size;
  const float* q = (const float*)d_in[0];
  const float* k = (const float*)d_in[1];
  const float* v = (const float*)d_in[2];
  const int*   m = (const int*)d_in[3];
  float*       o = (float*)d_out;
  dim3 grid(LQ_ / QTILE, B_);
  dim3 block(256);
  sigmoid_attn_kernel<<<grid, block, 0, stream>>>(q, k, v, m, o);
}
